// EGNN_19198503813802
// MI455X (gfx1250) — matrix operand order repaired and checked by reading, not compiled, not run on hardware
//
#include <hip/hip_runtime.h>
#include <hip/hip_bf16.h>
#include <stdint.h>

#define BATCH 2
#define NN 512
#define HH 128
#define GG 32
#define SPLIT 8          // sender-range splits per receiver tile

typedef __attribute__((ext_vector_type(16))) _Float16 v16h;
typedef __attribute__((ext_vector_type(8)))  float    v8f;

union ABu { v16h v; uint4 q[2]; };

__device__ __forceinline__ float siluf(float x){ return x / (1.0f + __expf(-x)); }
__device__ __forceinline__ float sigmf(float x){ return 1.0f / (1.0f + __expf(-x)); }

// ---------------------------------------------------------------------------
// WMMA GEMM: 16 edges (rows, in per-wave LDS f16 [16][128]) x W^T[128][128]
// (f16, global, row n holds K contiguous) -> acc[8] (8 N-tiles of 16 cols).
// A layout (16x32 f16): lanes 0-15 row m, halfs = K 0..7,16..23 ; lanes 16-31
// row m, halfs = K 8..15,24..31.  B layout (32x16): lanes 0-15 K 0..7,16..23 col n,
// lanes 16-31 K 8..15,24..31.  D: row = j + 8*(lane>=16), col = lane&15.
// ---------------------------------------------------------------------------
__device__ __forceinline__ void gemm16x128x128(const _Float16* Xw,
                                               const _Float16* __restrict__ WT,
                                               int lane, v8f acc[8]) {
  const int m  = lane & 15;
  #pragma unroll
  for (int kt = 0; kt < 4; ++kt) {
    ABu a;
    {
      const int lo = kt * 32 + ((lane & 16) ? 8 : 0);
      a.q[0] = *((const uint4*)(Xw + m * 128 + lo));
      a.q[1] = *((const uint4*)(Xw + m * 128 + lo + 16));
    }
    const int kb = kt * 32 + ((lane & 16) ? 8 : 0);
    #pragma unroll
    for (int nt = 0; nt < 8; ++nt) {
      ABu bm;
      const uint4* q = (const uint4*)(WT + (nt * 16 + m) * 128 + kb);
      bm.q[0] = q[0];
      bm.q[1] = q[2];
      acc[nt] = __builtin_amdgcn_wmma_f32_16x16x32_f16(
          false, a.v, false, bm.v, (short)0, acc[nt], false, false);
    }
  }
}

// ---------------------------------------------------------------------------
// Edge kernel: one block = 16 receivers x 64 senders (one of SPLIT ranges).
// 8 waves; each wave-chunk = 1 sender x 16 receivers = one 16-row M-tile.
// ---------------------------------------------------------------------------
struct EdgeArgs {
  const float* pos;
  const float* Hs;          // [B*N][128]  (includes b1)
  const float* Hr;          // [B*N][128]
  const _Float16* WT2e;     // phi_e L2  W^T f16 [128][128]
  const _Float16* WTx0;     // phi_x L1  W^T
  const _Float16* WTx1;     // phi_x L2  W^T
  const float* b2e;
  const float* bx0;
  const float* bx1;
  const float* wxo; const float* bxo;   // phi_x_out
  const float* wg;  const float* bg;    // e_gate
  const float* wl;                      // phi_e L1 row 256 (length weight)
  float* m_i;
  float* shift;
};

__launch_bounds__(256, 1)
__global__ void egnn_edge_kernel(EdgeArgs A) {
  __shared__ float HrT[16 * 128];
  __shared__ float smallv[6 * 128];   // b2e | bx0 | bx1 | wxo | wg | wl
  __shared__ __align__(16) _Float16 X[8][16 * 128];
  __shared__ float lenS[8][16];
  __shared__ float phixS[8][16];

  const int tid  = threadIdx.x;
  const int lane = tid & 31;
  const int wave = tid >> 5;
  const int b    = blockIdx.x / (32 * SPLIT);
  const int rt   = (blockIdx.x / SPLIT) & 31;
  const int sp   = blockIdx.x & (SPLIT - 1);
  const int rBase = rt * 16;

  #pragma unroll
  for (int i = 0; i < 8; ++i) {
    int idx = tid + i * 256;
    HrT[idx] = A.Hr[(size_t)(b * NN + rBase + (idx >> 7)) * 128 + (idx & 127)];
  }
  if (tid < 128) {
    smallv[tid]       = A.b2e[tid];
    smallv[128 + tid] = A.bx0[tid];
    smallv[256 + tid] = A.bx1[tid];
    smallv[384 + tid] = A.wxo[tid];
    smallv[512 + tid] = A.wg[tid];
    smallv[640 + tid] = A.wl[tid];
  }
  __syncthreads();

  const int n  = lane & 15;
  const int hb = lane >> 4;
  const float bgv  = A.bg[0];
  const float bxov = A.bxo[0];

  float macc[8][8];
  #pragma unroll
  for (int a = 0; a < 8; ++a)
    #pragma unroll
    for (int j = 0; j < 8; ++j) macc[a][j] = 0.f;
  float shx = 0.f, shy = 0.f, shz = 0.f;
  float vx = 0.f, vy = 0.f, vz = 0.f;

  _Float16* Xw = X[wave];

  for (int st = 0; st < 4; ++st)
  for (int c = 0; c < 2; ++c) {
    const int s = sp * 64 + st * 16 + c * 8 + wave;   // sender node (0..511)

    // geometry: lanes 0..15 own receiver rBase+lane
    if (lane < 16) {
      const float* pr = A.pos + (size_t)(b * NN + rBase + lane) * 3;
      const float* ps = A.pos + (size_t)(b * NN + s) * 3;
      vx = pr[0] - ps[0]; vy = pr[1] - ps[1]; vz = pr[2] - ps[2];
      float l2 = vx * vx + vy * vy + vz * vz;
      lenS[wave][lane] = (l2 > 0.f) ? sqrtf(l2) : 0.f;
    }

    // X0 = silu(Hs[s] + Hr[r] + len * wl)  -> per-wave LDS f16 [16][128]
    const float* HsRow = A.Hs + (size_t)(b * NN + s) * 128;
    #pragma unroll 4
    for (int i = 0; i < 64; ++i) {
      int idx = i * 32 + lane;
      int e = idx >> 7, f = idx & 127;
      float v = HsRow[f] + HrT[e * 128 + f] + lenS[wave][e] * smallv[640 + f];
      Xw[e * 128 + f] = (_Float16)siluf(v);
    }

    // ---- GEMM1: m_ij = silu(X0 @ W2e + b2e) --------------------------------
    v8f acc[8];
    #pragma unroll
    for (int nt = 0; nt < 8; ++nt)
      #pragma unroll
      for (int j = 0; j < 8; ++j) acc[nt][j] = 0.f;
    gemm16x128x128(Xw, A.WT2e, lane, acc);

    float gp[8];
    #pragma unroll
    for (int j = 0; j < 8; ++j) gp[j] = 0.f;
    #pragma unroll
    for (int nt = 0; nt < 8; ++nt) {
      float bb  = smallv[nt * 16 + n];
      float wgv = smallv[512 + nt * 16 + n];
      #pragma unroll
      for (int j = 0; j < 8; ++j) {
        float mv = siluf(acc[nt][j] + bb);
        acc[nt][j] = mv;
        gp[j] += mv * wgv;
      }
    }
    #pragma unroll
    for (int j = 0; j < 8; ++j)
      #pragma unroll
      for (int off = 1; off < 16; off <<= 1) gp[j] += __shfl_xor(gp[j], off, 32);

    float gate[8];
    #pragma unroll
    for (int j = 0; j < 8; ++j) {
      int r = rBase + j + 8 * hb;
      gate[j] = (s == r) ? 0.f : sigmf(gp[j] + bgv);   // mask self-edges
    }
    #pragma unroll
    for (int nt = 0; nt < 8; ++nt)
      #pragma unroll
      for (int j = 0; j < 8; ++j) macc[nt][j] += acc[nt][j] * gate[j];
    #pragma unroll
    for (int nt = 0; nt < 8; ++nt)
      #pragma unroll
      for (int j = 0; j < 8; ++j)
        Xw[(j + 8 * hb) * 128 + nt * 16 + n] = (_Float16)acc[nt][j];

    // ---- GEMM2: t1 = silu(m @ Wx0 + bx0) -----------------------------------
    #pragma unroll
    for (int nt = 0; nt < 8; ++nt)
      #pragma unroll
      for (int j = 0; j < 8; ++j) acc[nt][j] = 0.f;
    gemm16x128x128(Xw, A.WTx0, lane, acc);
    #pragma unroll
    for (int nt = 0; nt < 8; ++nt) {
      float bb = smallv[128 + nt * 16 + n];
      #pragma unroll
      for (int j = 0; j < 8; ++j)
        Xw[(j + 8 * hb) * 128 + nt * 16 + n] = (_Float16)siluf(acc[nt][j] + bb);
    }

    // ---- GEMM3: t2 = silu(t1 @ Wx1 + bx1); phix = t2 . wxo + bxo -----------
    #pragma unroll
    for (int nt = 0; nt < 8; ++nt)
      #pragma unroll
      for (int j = 0; j < 8; ++j) acc[nt][j] = 0.f;
    gemm16x128x128(Xw, A.WTx1, lane, acc);

    float pp[8];
    #pragma unroll
    for (int j = 0; j < 8; ++j) pp[j] = 0.f;
    #pragma unroll
    for (int nt = 0; nt < 8; ++nt) {
      float bb = smallv[256 + nt * 16 + n];
      float wo = smallv[384 + nt * 16 + n];
      #pragma unroll
      for (int j = 0; j < 8; ++j) pp[j] += siluf(acc[nt][j] + bb) * wo;
    }
    #pragma unroll
    for (int j = 0; j < 8; ++j)
      #pragma unroll
      for (int off = 1; off < 16; off <<= 1) pp[j] += __shfl_xor(pp[j], off, 32);
    #pragma unroll
    for (int j = 0; j < 8; ++j)
      if (n == j) phixS[wave][hb * 8 + j] = pp[j] + bxov;

    if (lane < 16) {
      float ph  = phixS[wave][lane];
      float w   = ph / (1.0f + lenS[wave][lane]);
      shx += w * vx; shy += w * vy; shz += w * vz;
    }
  }

  // flush accumulators
  #pragma unroll
  for (int nt = 0; nt < 8; ++nt)
    #pragma unroll
    for (int j = 0; j < 8; ++j)
      atomicAdd(&A.m_i[(size_t)(b * NN + rBase + j + 8 * hb) * 128 + nt * 16 + n],
                macc[nt][j]);
  if (lane < 16) {
    float* sh = A.shift + (size_t)(b * NN + rBase + lane) * 3;
    atomicAdd(sh + 0, shx);
    atomicAdd(sh + 1, shy);
    atomicAdd(sh + 2, shz);
  }
}

// ---------------------------------------------------------------------------
// Small node-level kernels (fp32; negligible FLOPs vs edge GEMMs)
// ---------------------------------------------------------------------------
__global__ void egnn_mean(const float* __restrict__ pos, float* mean0) {
  int t = threadIdx.x;
  if (t >= BATCH * 3) return;
  int b = t / 3, c = t % 3;
  float s = 0.f;
  for (int i = 0; i < NN; ++i) s += pos[(size_t)(b * NN + i) * 3 + c];
  mean0[t] = s / (float)NN;
}

__global__ void egnn_wconv(const float* __restrict__ W, _Float16* WT) {
  int t = blockIdx.x * blockDim.x + threadIdx.x;   // 16384
  int nn_ = t >> 7, k = t & 127;
  WT[t] = (_Float16)W[k * 128 + nn_];
}

__global__ void egnn_embed(const float* __restrict__ h, const float* __restrict__ g,
                           const float* __restrict__ W, const float* __restrict__ bb,
                           float* he) {
  int t = blockIdx.x * blockDim.x + threadIdx.x;   // B*N*128
  int f = t & 127, bn = t >> 7, bidx = bn >> 9;
  const float* hr = h + (size_t)bn * 128;
  const float* gr = g + bidx * GG;
  float acc = bb[f];
  for (int k = 0; k < 128; ++k) acc += hr[k] * W[k * 128 + f];
  for (int k = 0; k < GG; ++k)  acc += gr[k] * W[(128 + k) * 128 + f];
  he[t] = acc;
}

__global__ void egnn_hshr(const float* __restrict__ he, const float* __restrict__ W1,
                          const float* __restrict__ b1, float* Hs, float* Hr) {
  int t = blockIdx.x * blockDim.x + threadIdx.x;
  int f = t & 127, bn = t >> 7;
  const float* hr = he + (size_t)bn * 128;
  float s = b1[f], r = 0.f;
  for (int k = 0; k < 128; ++k) {
    float hv = hr[k];
    s += hv * W1[k * 128 + f];
    r += hv * W1[(128 + k) * 128 + f];
  }
  Hs[t] = s;
  Hr[t] = r;
}

__global__ void egnn_ph1(const float* __restrict__ m_i, const float* __restrict__ he,
                         const float* __restrict__ W, const float* __restrict__ bb,
                         float* t1) {
  int t = blockIdx.x * blockDim.x + threadIdx.x;
  int f = t & 127, bn = t >> 7;
  const float invs = 1.0f / sqrtf((float)(NN - 1));
  const float* mr = m_i + (size_t)bn * 128;
  const float* hr = he + (size_t)bn * 128;
  float acc = bb[f];
  for (int k = 0; k < 128; ++k) acc += (mr[k] * invs) * W[k * 128 + f];
  for (int k = 0; k < 128; ++k) acc += hr[k] * W[(128 + k) * 128 + f];
  t1[t] = siluf(acc);
}

__global__ void egnn_dense_silu(const float* __restrict__ in, const float* __restrict__ W,
                                const float* __restrict__ bb, float* out) {
  int t = blockIdx.x * blockDim.x + threadIdx.x;
  int f = t & 127, bn = t >> 7;
  const float* xr = in + (size_t)bn * 128;
  float acc = bb[f];
  for (int k = 0; k < 128; ++k) acc += xr[k] * W[k * 128 + f];
  out[t] = siluf(acc);
}

__global__ void egnn_ph3(const float* __restrict__ t2, const float* __restrict__ W,
                         const float* __restrict__ bb, const float* __restrict__ he,
                         float* h) {
  int t = blockIdx.x * blockDim.x + threadIdx.x;
  int f = t & 127, bn = t >> 7;
  const float* xr = t2 + (size_t)bn * 128;
  float acc = bb[f];
  for (int k = 0; k < 128; ++k) acc += xr[k] * W[k * 128 + f];
  h[t] = acc + he[t];   // residual
}

__global__ void egnn_posupd(float* pos, const float* __restrict__ shift) {
  int t = blockIdx.x * blockDim.x + threadIdx.x;
  if (t < BATCH * NN * 3) pos[t] += shift[t] * (1.0f / (float)(NN - 1));
}

__global__ void egnn_out(const float* __restrict__ h, const float* __restrict__ Wout,
                         const float* __restrict__ bout, const float* __restrict__ pos,
                         const float* __restrict__ pos0, const float* __restrict__ mean0,
                         float* out) {
  int t = blockIdx.x * blockDim.x + threadIdx.x;
  if (t >= BATCH * NN) return;
  int bidx = t >> 9;
  float sc = bout[0];
  const float* hr = h + (size_t)t * 128;
  for (int k = 0; k < 128; ++k) sc += hr[k] * Wout[k];
  for (int c = 0; c < 3; ++c)
    out[(size_t)t * 3 + c] = (pos[(size_t)t * 3 + c] - pos0[(size_t)t * 3 + c]
                              - mean0[bidx * 3 + c]) * sc;
}

// ---------------------------------------------------------------------------
extern "C" void kernel_launch(void* const* d_in, const int* in_sizes, int n_in,
                              void* d_out, int out_size, void* d_ws, size_t ws_size,
                              hipStream_t stream) {
  (void)in_sizes; (void)n_in; (void)out_size; (void)ws_size;
  const float* positions     = (const float*)d_in[0];
  const float* node_features = (const float*)d_in[1];
  const float* gfeat         = (const float*)d_in[2];
  auto P = [&](int i) { return (const float*)d_in[i]; };

  char* ws = (char*)d_ws;
  size_t off = 0;
  auto alloc = [&](size_t bytes) -> char* {
    char* p = ws + off;
    off += (bytes + 255) & ~(size_t)255;
    return p;
  };
  const size_t NHbytes = (size_t)BATCH * NN * HH * sizeof(float);
  float* h_cur   = (float*)alloc(NHbytes);
  float* h_emb   = (float*)alloc(NHbytes);
  float* Hs      = (float*)alloc(NHbytes);
  float* Hr      = (float*)alloc(NHbytes);
  float* m_i     = (float*)alloc(NHbytes);
  float* t1      = (float*)alloc(NHbytes);
  float* t2      = (float*)alloc(NHbytes);
  float* pos_cur = (float*)alloc((size_t)BATCH * NN * 3 * sizeof(float));
  float* shiftb  = (float*)alloc((size_t)BATCH * NN * 3 * sizeof(float));
  float* mean0   = (float*)alloc(256);
  _Float16* WT2e = (_Float16*)alloc(128 * 128 * sizeof(_Float16));
  _Float16* WTx0 = (_Float16*)alloc(128 * 128 * sizeof(_Float16));
  _Float16* WTx1 = (_Float16*)alloc(128 * 128 * sizeof(_Float16));

  hipMemcpyAsync(pos_cur, positions, (size_t)BATCH * NN * 3 * sizeof(float),
                 hipMemcpyDeviceToDevice, stream);
  hipMemcpyAsync(h_cur, node_features, NHbytes, hipMemcpyDeviceToDevice, stream);
  egnn_mean<<<1, 32, 0, stream>>>(positions, mean0);

  for (int bl = 0; bl < 2; ++bl) {
    const int base = 3 + bl * 20;
    const float* We  = P(base + 0);  const float* be  = P(base + 1);
    const float* W1  = P(base + 2);  const float* b1  = P(base + 3);
    const float* W2e = P(base + 4);  const float* b2e = P(base + 5);
    const float* Wx0 = P(base + 6);  const float* bx0 = P(base + 7);
    const float* Wx1 = P(base + 8);  const float* bx1 = P(base + 9);
    const float* Wxo = P(base + 10); const float* bxo = P(base + 11);
    const float* Wg  = P(base + 12); const float* bg  = P(base + 13);
    const float* Wh0 = P(base + 14); const float* bh0 = P(base + 15);
    const float* Wh1 = P(base + 16); const float* bh1 = P(base + 17);
    const float* Wh2 = P(base + 18); const float* bh2 = P(base + 19);

    egnn_wconv<<<64, 256, 0, stream>>>(W2e, WT2e);
    egnn_wconv<<<64, 256, 0, stream>>>(Wx0, WTx0);
    egnn_wconv<<<64, 256, 0, stream>>>(Wx1, WTx1);
    egnn_embed<<<512, 256, 0, stream>>>(h_cur, gfeat, We, be, h_emb);
    egnn_hshr<<<512, 256, 0, stream>>>(h_emb, W1, b1, Hs, Hr);
    hipMemsetAsync(m_i, 0, NHbytes, stream);
    hipMemsetAsync(shiftb, 0, (size_t)BATCH * NN * 3 * sizeof(float), stream);

    EdgeArgs ea;
    ea.pos = pos_cur; ea.Hs = Hs; ea.Hr = Hr;
    ea.WT2e = WT2e; ea.WTx0 = WTx0; ea.WTx1 = WTx1;
    ea.b2e = b2e; ea.bx0 = bx0; ea.bx1 = bx1;
    ea.wxo = Wxo; ea.bxo = bxo; ea.wg = Wg; ea.bg = bg;
    ea.wl = W1 + 256 * 128;
    ea.m_i = m_i; ea.shift = shiftb;
    egnn_edge_kernel<<<BATCH * 32 * SPLIT, 256, 0, stream>>>(ea);

    egnn_posupd<<<12, 256, 0, stream>>>(pos_cur, shiftb);
    egnn_ph1<<<512, 256, 0, stream>>>(m_i, h_emb, Wh0, bh0, t1);
    egnn_dense_silu<<<512, 256, 0, stream>>>(t1, Wh1, bh1, t2);
    egnn_ph3<<<512, 256, 0, stream>>>(t2, Wh2, bh2, h_emb, h_cur);
  }

  const float* Wout = P(43);
  const float* bout = P(44);
  egnn_out<<<8, 128, 0, stream>>>(h_cur, Wout, bout, pos_cur, positions, mean0,
                                  (float*)d_out);
}
